// LSTM_87995289961163
// MI455X (gfx1250) — matrix operand order repaired and checked by reading, not compiled, not run on hardware
//
#include <hip/hip_runtime.h>
#include <cstddef>

// ---------------- types matching CDNA5 WMMA operand layouts ----------------
typedef __attribute__((ext_vector_type(16))) __bf16        bf16x16;
typedef __attribute__((ext_vector_type(8)))  float         f32x8;
typedef __attribute__((ext_vector_type(4)))  unsigned int  u32x4;

struct Frag { u32x4 lo, hi; };   // 32 bytes == 16 bf16 == one WMMA A/B fragment

#define T_STEPS 256
#define BATCH   128
#define DIN     1024
#define UHID    1024
#define K2      2048          // DIN + UHID (fused GEMM K)
#define NZ      4096          // 4*UHID gate columns
#define BU      (BATCH*UHID)

static __device__ __forceinline__ unsigned short f2bf(float f) {
  unsigned int x = __builtin_bit_cast(unsigned int, f);
  x += 0x7FFFu + ((x >> 16) & 1u);          // round-to-nearest-even
  return (unsigned short)(x >> 16);
}

static __device__ __forceinline__ u32x4 ld16(const unsigned short* p) {
  return *reinterpret_cast<const u32x4*>(p); // global_load_b128
}

// Fast gate nonlinearities on the hardware transcendental pipe:
//   v_exp_f32 + v_rcp_f32 instead of libm tanhf / IEEE division.
static __device__ __forceinline__ float fsigmoid(float x) {
  return __builtin_amdgcn_rcpf(1.0f + __expf(-x));
}
static __device__ __forceinline__ float ftanh(float x) {
  return 1.0f - 2.0f * __builtin_amdgcn_rcpf(1.0f + __expf(2.0f * x));
}

static __device__ __forceinline__ f32x8 wmma_bf16(bf16x16 a, bf16x16 b, f32x8 c) {
  return __builtin_amdgcn_wmma_f32_16x16x32_bf16(false, a, false, b,
                                                 (short)0, c, false, false);
}

// ---------------- one-time prep kernels ----------------
// X fp32 -> bf16
__global__ __launch_bounds__(256) void convert_x(const float* __restrict__ X,
                                                 unsigned short* __restrict__ Xbf) {
  size_t i = (size_t)blockIdx.x * 256 + threadIdx.x;
  Xbf[i] = f2bf(X[i]);
}

// Combined transposed weights: WT[n][k] = (k<DIN ? Wx[k][n] : Wh[k-DIN][n]), bf16
__global__ __launch_bounds__(256) void convert_w(const float* __restrict__ Wx,
                                                 const float* __restrict__ Wh,
                                                 unsigned short* __restrict__ WT) {
  size_t i = (size_t)blockIdx.x * 256 + threadIdx.x;   // 0 .. NZ*K2-1
  int n = (int)(i >> 11);          // output column (0..4095)
  int k = (int)(i & (K2 - 1));     // reduction index (0..2047)
  float v = (k < DIN) ? Wx[(size_t)k * NZ + n]
                      : Wh[(size_t)(k - DIN) * NZ + n];
  WT[i] = f2bf(v);
}

// c0 / h0 from the "states" input
__global__ __launch_bounds__(256) void init_state(const float* __restrict__ S,
                                                  float* __restrict__ Cws,
                                                  unsigned short* __restrict__ H0) {
  int i = blockIdx.x * 256 + threadIdx.x;   // 0 .. BU-1
  Cws[i] = S[i];                            // c0
  H0[i]  = f2bf(S[BU + i]);                 // h0
}

// ---------------- per-timestep fused GEMM + gates kernel ----------------
// Grid: (16, 4), block = 128 threads = 4 waves. Each wave owns:
//   2 batch-row tiles (32 rows) x 4 gate tiles over the same 16 hidden columns
//   => 8 independent WMMA accumulator chains, A reused x4, B reused x2,
//      and the whole LSTM cell epilogue is wave-local register math.
__global__ __launch_bounds__(128) void lstm_step(
    int t,
    const unsigned short* __restrict__ Xbf,
    const unsigned short* __restrict__ WT,
    const unsigned short* __restrict__ Hread,
    unsigned short*       __restrict__ Hwrite,
    float*                __restrict__ Cws,
    const int*            __restrict__ masks,
    const float*          __restrict__ bias,
    float*                __restrict__ Out) {
  const int tid  = threadIdx.x;
  const int lane = tid & 31;
  const int wq   = tid >> 5;                      // column tile within block
  const int r    = lane & 15;                     // A row / WT row (output col)
  const int hb   = lane >> 4;                     // K-half select (WMMA layout)
  const int m0   = blockIdx.y << 5;               // 32 batch rows per wave
  const int jb   = ((blockIdx.x << 2) + wq) << 4; // hidden-column tile base

  const unsigned short* A0 = Xbf + ((size_t)t * BATCH + m0 + r) * DIN;
  const unsigned short* A1 = A0 + (size_t)16 * DIN;
  const unsigned short* H0 = Hread + (size_t)(m0 + r) * UHID;
  const unsigned short* H1 = H0 + (size_t)16 * UHID;
  const unsigned short* Bp0 = WT + (size_t)(0 * UHID + jb + r) * K2;
  const unsigned short* Bp1 = WT + (size_t)(1 * UHID + jb + r) * K2;
  const unsigned short* Bp2 = WT + (size_t)(2 * UHID + jb + r) * K2;
  const unsigned short* Bp3 = WT + (size_t)(3 * UHID + jb + r) * K2;

  const bool kill0 = masks[t * BATCH + m0 + r] != 0;        // m = 1-mask
  const bool kill1 = masks[t * BATCH + m0 + 16 + r] != 0;

  f32x8 acc[2][4] = {};

  // ---- K sweep over input contribution (x_t) ----
  for (int k0 = 0; k0 < DIN; k0 += 32) {
    Frag a0, a1;
    a0.lo = ld16(A0 + k0 + hb * 8);
    a0.hi = ld16(A0 + k0 + 16 + hb * 8);
    a1.lo = ld16(A1 + k0 + hb * 8);
    a1.hi = ld16(A1 + k0 + 16 + hb * 8);
    const bf16x16 av0 = __builtin_bit_cast(bf16x16, a0);
    const bf16x16 av1 = __builtin_bit_cast(bf16x16, a1);
    const unsigned short* bp[4] = {Bp0, Bp1, Bp2, Bp3};
#pragma unroll
    for (int q = 0; q < 4; ++q) {
      Frag b;
      b.lo = ld16(bp[q] + k0 + hb * 8);
      b.hi = ld16(bp[q] + k0 + 16 + hb * 8);
      const bf16x16 bv = __builtin_bit_cast(bf16x16, b);
      acc[0][q] = wmma_bf16(av0, bv, acc[0][q]);
      acc[1][q] = wmma_bf16(av1, bv, acc[1][q]);
    }
  }

  // ---- K sweep over recurrent contribution (m * h_{t-1}) ----
  const u32x4 zero4 = {0u, 0u, 0u, 0u};
  for (int k0 = 0; k0 < UHID; k0 += 32) {
    Frag a0, a1;
    u32x4 l00 = ld16(H0 + k0 + hb * 8);
    u32x4 l01 = ld16(H0 + k0 + 16 + hb * 8);
    u32x4 l10 = ld16(H1 + k0 + hb * 8);
    u32x4 l11 = ld16(H1 + k0 + 16 + hb * 8);
    a0.lo = kill0 ? zero4 : l00;           // per-row state mask -> v_cndmask
    a0.hi = kill0 ? zero4 : l01;
    a1.lo = kill1 ? zero4 : l10;
    a1.hi = kill1 ? zero4 : l11;
    const bf16x16 av0 = __builtin_bit_cast(bf16x16, a0);
    const bf16x16 av1 = __builtin_bit_cast(bf16x16, a1);
    const unsigned short* bp[4] = {Bp0, Bp1, Bp2, Bp3};
#pragma unroll
    for (int q = 0; q < 4; ++q) {
      Frag b;
      b.lo = ld16(bp[q] + DIN + k0 + hb * 8);
      b.hi = ld16(bp[q] + DIN + k0 + 16 + hb * 8);
      const bf16x16 bv = __builtin_bit_cast(bf16x16, b);
      acc[0][q] = wmma_bf16(av0, bv, acc[0][q]);
      acc[1][q] = wmma_bf16(av1, bv, acc[1][q]);
    }
  }

  // ---- wave-local LSTM cell epilogue, directly in the C/D register layout ----
  // C/D layout: column n = lane&15, row = (lane>>4)*8 + v
  const int j = jb + r;                     // hidden unit (column) of this lane
  const float bI = bias[j];
  const float bF = bias[j + UHID];
  const float bO = bias[j + 2 * UHID];
  const float bU = bias[j + 3 * UHID];

#pragma unroll
  for (int rb = 0; rb < 2; ++rb) {
#pragma unroll
    for (int v = 0; v < 8; ++v) {
      const int brow = m0 + rb * 16 + hb * 8 + v;           // batch row
      const float mval = masks[t * BATCH + brow] ? 0.0f : 1.0f;
      const size_t ci = (size_t)brow * UHID + j;

      const float zi = acc[rb][0][v] + bI;
      const float zf = acc[rb][1][v] + bF;
      const float zo = acc[rb][2][v] + bO;
      const float zu = acc[rb][3][v] + bU;

      const float c  = Cws[ci] * mval;
      const float ig = fsigmoid(zi);
      const float fg = fsigmoid(zf);
      const float og = fsigmoid(zo);
      const float ug = ftanh(zu);
      const float cn = fg * c + ig * ug;
      const float h  = og * ftanh(cn);

      Cws[ci] = cn;
      Out[(size_t)t * BU + ci] = h;
      Hwrite[ci] = f2bf(h);
    }
  }
}

// ---------------- final states ----------------
__global__ __launch_bounds__(256) void finalize(const float* __restrict__ Cws,
                                                float* __restrict__ Out) {
  int i = blockIdx.x * 256 + threadIdx.x;    // 0 .. BU-1
  const size_t base = (size_t)T_STEPS * BU;
  Out[base + i]      = Cws[i];                               // final c
  Out[base + BU + i] = Out[(size_t)(T_STEPS - 1) * BU + i];  // final h = last output
}

// ---------------- launch ----------------
extern "C" void kernel_launch(void* const* d_in, const int* in_sizes, int n_in,
                              void* d_out, int out_size, void* d_ws, size_t ws_size,
                              hipStream_t stream) {
  const float* X     = (const float*)d_in[0];  // [T,B,D]
  const float* S     = (const float*)d_in[1];  // [2,B,U]
  const int*   masks = (const int*)  d_in[2];  // [T,B]
  const float* Wx    = (const float*)d_in[3];  // [D,4U]
  const float* Wh    = (const float*)d_in[4];  // [U,4U]
  const float* bias  = (const float*)d_in[5];  // [4U]
  float*       out   = (float*)d_out;          // [T,B,U] then [2,B,U]

  char* ws = (char*)d_ws;
  unsigned short* Xbf = (unsigned short*)(ws);                 // 67,108,864 B
  unsigned short* WT  = (unsigned short*)(ws + 67108864);      // 16,777,216 B
  float*          Cws = (float*)         (ws + 83886080);      //    524,288 B
  unsigned short* hA  = (unsigned short*)(ws + 84410368);      //    262,144 B
  unsigned short* hB  = (unsigned short*)(ws + 84672512);      //    262,144 B

  convert_x<<<(T_STEPS * BATCH * DIN) / 256, 256, 0, stream>>>(X, Xbf);
  convert_w<<<(NZ * K2) / 256,               256, 0, stream>>>(Wx, Wh, WT);
  init_state<<<BU / 256,                     256, 0, stream>>>(S, Cws, hA);

  for (int t = 0; t < T_STEPS; ++t) {
    unsigned short* hr = (t & 1) ? hB : hA;   // h_{t-1} (ping-pong: no races)
    unsigned short* hw = (t & 1) ? hA : hB;   // h_t
    lstm_step<<<dim3(16, 4), 128, 0, stream>>>(
        t, Xbf, WT, hr, hw, Cws, masks, bias, out);
  }

  finalize<<<BU / 256, 256, 0, stream>>>(Cws, out);
}
